// CausalSelfAttention_12266426597787
// MI455X (gfx1250) — matrix operand order repaired and checked by reading, not compiled, not run on hardware
//
#include <hip/hip_runtime.h>

// Problem constants (B=4, T=2048, C=1024, H=16, D=64)
#define B_  4
#define T_  2048
#define C_  1024
#define H_  16
#define D_  64
#define N3_ 3072
#define M_  (B_ * T_)   // 8192

typedef float  v4f   __attribute__((ext_vector_type(4)));
typedef float  v8f   __attribute__((ext_vector_type(8)));
typedef float  v16f  __attribute__((ext_vector_type(16)));
typedef __bf16 v4bf  __attribute__((ext_vector_type(4)));
typedef __bf16 v8bf  __attribute__((ext_vector_type(8)));
typedef __bf16 v16bf __attribute__((ext_vector_type(16)));

// bf16 carried as ushort across the host/kernel ABI (avoids host-side __bf16)
typedef unsigned short bf16s;

// Hardware bf16 convert (backend emits v_cvt_pk_bf16_f32)
static __device__ __forceinline__ __bf16 bf16c(float f) { return (__bf16)f; }

// 16 contiguous bf16 (32B) -> one A/B fragment register group
static __device__ __forceinline__ v16bf ldg16(const __bf16* p) {
  v8bf lo = *(const v8bf*)p;
  v8bf hp = *(const v8bf*)(p + 8);
  v16bf r;
#pragma unroll
  for (int j = 0; j < 8; ++j) { r[j] = lo[j]; r[j + 8] = hp[j]; }
  return r;
}

// ---- CDNA5 async global->LDS copies (ASYNCcnt path, no VGPR round trip) ----
static __device__ __forceinline__ void async_b128(unsigned lds, const void* g) {
  asm volatile("global_load_async_to_lds_b128 %0, %1, off"
               :: "v"(lds), "v"((unsigned long long)(size_t)g) : "memory");
}
template <int N>
static __device__ __forceinline__ void wait_async_n() {
  asm volatile("s_wait_asynccnt %0" :: "i"(N) : "memory");
}
template <typename Tp>
static __device__ __forceinline__ unsigned lds_addr(Tp* p) {
  return (unsigned)(size_t)p;
}

// ---------------------------------------------------------------------------
// Kernel 1: qkv = x @ w_attn + b_attn  (M=8192, N=3072, K=1024)
// Block tile 64(M) x 128(N), 8 waves: wave = 4 Mtiles x 1 Ntile (4 WMMA/kstep).
// Double-buffered LDS; B via async DMA (in-order ASYNCcnt pipelining),
// A split-staged (loads at top, cvt+store after compute).
// Writes Q [B,H,T,D] (scaled by 1/sqrt(D)*log2e), K [B,H,T,D], V^T [B,H,D,T].
// ---------------------------------------------------------------------------
__global__ __launch_bounds__(256) void qkv_gemm_kernel(
    const float* __restrict__ x, const float* __restrict__ w,
    const float* __restrict__ bias,
    bf16s* __restrict__ qo, bf16s* __restrict__ ko, bf16s* __restrict__ vto) {
  __shared__ __bf16 As[2][64][40];    // 80B pitch: 16B-aligned, conflict-free
  __shared__ float  Bsf[2][32][128];  // [k][n] raw f32, filled by async DMA

  const int tid  = threadIdx.x;
  const int lane = tid & 31;
  const int wave = tid >> 5;
  const int m0 = blockIdx.y << 6;
  const int nb = blockIdx.x << 7;
  const int r  = lane & 15;
  const int hi = lane >> 4;

  const int ae = tid << 3;                 // A staging: 8 f32 / thread
  const int ar = ae >> 5, ac = ae & 31;
  const int br = tid >> 3;                 // B staging: k-row, 16 f32 / thread
  const int bc = (tid & 7) << 4;

  v8f acc[4] = {};

  // ---- prologue: fill buffer 0 ----
  {
    const float* wp = w + (size_t)br * N3_ + (nb + bc);
    const unsigned lb = lds_addr(&Bsf[0][br][bc]);
#pragma unroll
    for (int c = 0; c < 4; ++c) async_b128(lb + 16u * c, wp + 4 * c);
    const float* xp = x + (size_t)(m0 + ar) * C_ + ac;
    *(v4bf*)&As[0][ar][ac]     = __builtin_convertvector(*(const v4f*)xp, v4bf);
    *(v4bf*)&As[0][ar][ac + 4] = __builtin_convertvector(*(const v4f*)(xp + 4), v4bf);
  }

  for (int k0 = 0; k0 < C_; k0 += 32) {
    const int cur = (k0 >> 5) & 1;
    const bool hasNext = (k0 + 32) < C_;
    v4f xa0, xa1;
    if (hasNext) {                       // issue next tile into other buffer
      const float* wp = w + (size_t)(k0 + 32 + br) * N3_ + (nb + bc);
      const unsigned lb = lds_addr(&Bsf[1 - cur][br][bc]);
#pragma unroll
      for (int c = 0; c < 4; ++c) async_b128(lb + 16u * c, wp + 4 * c);
      const float* xp = x + (size_t)(m0 + ar) * C_ + (k0 + 32 + ac);
      xa0 = *(const v4f*)xp;
      xa1 = *(const v4f*)(xp + 4);
      wait_async_n<4>();                 // in-order: current buffer resident
    } else {
      wait_async_n<0>();
    }
    __syncthreads();

    // ---- compute from current buffer: 1 B-frag reused by 4 M-tiles ----
    v16bf bfr;
    {
      v16f bt;
#pragma unroll
      for (int j = 0; j < 16; ++j) bt[j] = Bsf[cur][hi * 16 + j][(wave << 4) + r];
      bfr = __builtin_convertvector(bt, v16bf);
    }
#pragma unroll
    for (int mt = 0; mt < 4; ++mt) {
      v8bf lo = *(const v8bf*)&As[cur][mt * 16 + r][hi * 16];
      v8bf hp = *(const v8bf*)&As[cur][mt * 16 + r][hi * 16 + 8];
      v16bf a;
#pragma unroll
      for (int j = 0; j < 8; ++j) { a[j] = lo[j]; a[j + 8] = hp[j]; }
      acc[mt] = __builtin_amdgcn_wmma_f32_16x16x32_bf16(false, a, false, bfr,
                                                        (short)0, acc[mt], false, false);
    }
    if (hasNext) {                       // late A store into other buffer
      *(v4bf*)&As[1 - cur][ar][ac]     = __builtin_convertvector(xa0, v4bf);
      *(v4bf*)&As[1 - cur][ar][ac + 4] = __builtin_convertvector(xa1, v4bf);
    }
    __syncthreads();
  }

  const int n  = nb + (wave << 4) + r;
  const float bv = bias[n];
  const int sect = n >> 10;        // 0=Q 1=K 2=V
  const int nc = n & (C_ - 1);
  const int h  = nc >> 6;
  const int d  = nc & (D_ - 1);
#pragma unroll
  for (int mt = 0; mt < 4; ++mt) {
#pragma unroll
    for (int i = 0; i < 8; ++i) {
      const int m = m0 + mt * 16 + i + (hi << 3);   // D-layout row
      const int bb = m >> 11;
      const int t  = m & (T_ - 1);
      float val = acc[mt][i] + bv;
      const size_t bh = (size_t)bb * H_ + h;
      if (sect == 0) {
        val *= 0.125f * 1.44269504088896340736f;    // 1/sqrt(64) * log2(e)
        qo[(bh * T_ + t) * D_ + d] = __builtin_bit_cast(unsigned short, bf16c(val));
      } else if (sect == 1) {
        ko[(bh * T_ + t) * D_ + d] = __builtin_bit_cast(unsigned short, bf16c(val));
      } else {
        vto[(bh * D_ + d) * T_ + t] = __builtin_bit_cast(unsigned short, bf16c(val));
      }
    }
  }
}

// ---------------------------------------------------------------------------
// Kernel 2: causal flash attention. One wave per (b, h, 16-row q tile).
// 64-key blocks: S = Q K^T (8 WMMAs), online softmax in exp2 domain,
// P via per-wave LDS to A-frag layout, O += P V (8 WMMAs).
// ---------------------------------------------------------------------------
__global__ __launch_bounds__(256) void attn_kernel(
    const bf16s* __restrict__ qs, const bf16s* __restrict__ ks,
    const bf16s* __restrict__ vts, bf16s* __restrict__ ys) {
  __shared__ __bf16 Pl[8][16][72];           // 144B pitch: aligned, conflict-free

  const int lane = threadIdx.x & 31;
  const int wave = threadIdx.x >> 5;
  const int wid  = (blockIdx.x << 3) + wave; // 0..8191
  const int qt = wid & 127;
  const int bh = wid >> 7;
  const int h  = bh & (H_ - 1);
  const int b  = bh >> 4;
  const int m0 = qt << 4;
  const int r  = lane & 15;
  const int hi = lane >> 4;

  const __bf16* qb = (const __bf16*)qs + (size_t)bh * T_ * D_;
  const __bf16* kb = (const __bf16*)ks + (size_t)bh * T_ * D_;
  const __bf16* vb = (const __bf16*)vts + (size_t)bh * D_ * T_;

  // Q A-frags for the two K=32 steps over D=64, kept resident
  v16bf qa[2];
#pragma unroll
  for (int s = 0; s < 2; ++s) {
    const __bf16* row = qb + (size_t)(m0 + r) * D_ + s * 32;
    v8bf lo = *(const v8bf*)(row + hi * 16);
    v8bf hp = *(const v8bf*)(row + hi * 16 + 8);
#pragma unroll
    for (int j = 0; j < 8; ++j) { qa[s][j] = lo[j]; qa[s][j + 8] = hp[j]; }
  }

  v8f o[4] = {};                    // 16x64 accumulator, 4 N-tiles
  float mrow[8], lrow[8];
#pragma unroll
  for (int i = 0; i < 8; ++i) { mrow[i] = -3.0e37f; lrow[i] = 0.0f; }

  for (int tb = 0; tb <= m0 + 15; tb += 64) {
    if (tb + 64 <= m0 + 15) {       // prefetch next K/V block
      __builtin_prefetch(kb + (size_t)(tb + 64 + r) * D_, 0, 1);
      __builtin_prefetch(vb + (size_t)r * T_ + tb + 64, 0, 1);
    }
    // ---- S = Q K^T over 64 keys (four 16x16 tiles) ----
    v8f s[4] = {};
#pragma unroll
    for (int ks2 = 0; ks2 < 2; ++ks2) {
#pragma unroll
      for (int nt = 0; nt < 4; ++nt) {
        v16bf bk = ldg16(kb + (size_t)(tb + nt * 16 + r) * D_ + ks2 * 32 + hi * 16);
        s[nt] = __builtin_amdgcn_wmma_f32_16x16x32_bf16(false, qa[ks2], false, bk,
                                                        (short)0, s[nt], false, false);
      }
    }
    // ---- causal mask (diagonal blocks only) ----
    if (tb + 63 > m0) {
#pragma unroll
      for (int nt = 0; nt < 4; ++nt)
#pragma unroll
        for (int i = 0; i < 8; ++i)
          if (tb + nt * 16 + r > m0 + i + (hi << 3)) s[nt][i] = -3.0e37f;
    }
    // ---- online softmax (log2e folded into Q; hardware v_exp) ----
#pragma unroll
    for (int i = 0; i < 8; ++i) {
      float mx = fmaxf(fmaxf(s[0][i], s[1][i]), fmaxf(s[2][i], s[3][i]));
#pragma unroll
      for (int off = 1; off < 16; off <<= 1)
        mx = fmaxf(mx, __shfl_xor(mx, off, 32));
      const float mnew  = fmaxf(mrow[i], mx);
      const float alpha = __builtin_amdgcn_exp2f(mrow[i] - mnew);
      const float p0 = __builtin_amdgcn_exp2f(s[0][i] - mnew);
      const float p1 = __builtin_amdgcn_exp2f(s[1][i] - mnew);
      const float p2 = __builtin_amdgcn_exp2f(s[2][i] - mnew);
      const float p3 = __builtin_amdgcn_exp2f(s[3][i] - mnew);
      float ls = (p0 + p1) + (p2 + p3);
#pragma unroll
      for (int off = 1; off < 16; off <<= 1)
        ls += __shfl_xor(ls, off, 32);
      lrow[i] = lrow[i] * alpha + ls;
      mrow[i] = mnew;
#pragma unroll
      for (int dt = 0; dt < 4; ++dt) o[dt][i] *= alpha;
      const int row = i + (hi << 3);
      Pl[wave][row][r]      = bf16c(p0);   // D-layout -> LDS
      Pl[wave][row][16 + r] = bf16c(p1);
      Pl[wave][row][32 + r] = bf16c(p2);
      Pl[wave][row][48 + r] = bf16c(p3);
    }
    // ---- re-read P as two A-frags (same-wave LDS ops are in-order) ----
    v16bf pa0, pa1;
    {
      v8bf l0 = *(const v8bf*)&Pl[wave][r][hi * 16];
      v8bf h0 = *(const v8bf*)&Pl[wave][r][hi * 16 + 8];
      v8bf l1 = *(const v8bf*)&Pl[wave][r][32 + hi * 16];
      v8bf h1 = *(const v8bf*)&Pl[wave][r][40 + hi * 16];
#pragma unroll
      for (int j = 0; j < 8; ++j) {
        pa0[j] = l0[j]; pa0[j + 8] = h0[j];
        pa1[j] = l1[j]; pa1[j + 8] = h1[j];
      }
    }
    // ---- O += P V : V^T rows give contiguous B-frags ----
#pragma unroll
    for (int dt = 0; dt < 4; ++dt) {
      v16bf bv0 = ldg16(vb + (size_t)((dt << 4) + r) * T_ + tb + hi * 16);
      o[dt] = __builtin_amdgcn_wmma_f32_16x16x32_bf16(false, pa0, false, bv0,
                                                      (short)0, o[dt], false, false);
      v16bf bv1 = ldg16(vb + (size_t)((dt << 4) + r) * T_ + tb + 32 + hi * 16);
      o[dt] = __builtin_amdgcn_wmma_f32_16x16x32_bf16(false, pa1, false, bv1,
                                                      (short)0, o[dt], false, false);
    }
  }

  // ---- normalize and store y[B,T,C] as bf16 for the proj GEMM ----
  __bf16* yb = (__bf16*)ys;
#pragma unroll
  for (int i = 0; i < 8; ++i) {
    const float inv = __builtin_amdgcn_rcpf(lrow[i]);
    const int t = m0 + i + (hi << 3);
    const size_t rowbase = ((size_t)b * T_ + t) * C_ + h * D_;
#pragma unroll
    for (int dt = 0; dt < 4; ++dt)
      yb[rowbase + (dt << 4) + r] = bf16c(o[dt][i] * inv);
  }
}

// ---------------------------------------------------------------------------
// Kernel 3: out = y @ w_proj + b_proj  (M=8192, N=1024, K=1024), fp32 out.
// Block tile 64x128, double-buffered; A (bf16 copy) and B (f32) both async DMA.
// ---------------------------------------------------------------------------
__global__ __launch_bounds__(256) void proj_gemm_kernel(
    const bf16s* __restrict__ ys, const float* __restrict__ w,
    const float* __restrict__ bias, float* __restrict__ out) {
  __shared__ __bf16 As[2][64][40];
  __shared__ float  Bsf[2][32][128];

  const int tid  = threadIdx.x;
  const int lane = tid & 31;
  const int wave = tid >> 5;
  const int m0 = blockIdx.y << 6;
  const int nb = blockIdx.x << 7;
  const int r  = lane & 15;
  const int hi = lane >> 4;

  const int ae = tid << 3;                 // A: 8 bf16 (16B) per thread
  const int ar = ae >> 5, ac = ae & 31;
  const int br = tid >> 3;
  const int bc = (tid & 7) << 4;

  v8f acc[4] = {};

  auto issue = [&](int buf, int k0) {
    const float* wp = w + (size_t)(k0 + br) * C_ + (nb + bc);
    const unsigned lb = lds_addr(&Bsf[buf][br][bc]);
#pragma unroll
    for (int c = 0; c < 4; ++c) async_b128(lb + 16u * c, wp + 4 * c);
    async_b128(lds_addr(&As[buf][ar][ac]),
               ys + (size_t)(m0 + ar) * C_ + (k0 + ac));
  };

  issue(0, 0);   // prologue

  for (int k0 = 0; k0 < C_; k0 += 32) {
    const int cur = (k0 >> 5) & 1;
    const bool hasNext = (k0 + 32) < C_;
    if (hasNext) { issue(1 - cur, k0 + 32); wait_async_n<5>(); }
    else         { wait_async_n<0>(); }
    __syncthreads();

    v16bf bfr;
    {
      v16f bt;
#pragma unroll
      for (int j = 0; j < 16; ++j) bt[j] = Bsf[cur][hi * 16 + j][(wave << 4) + r];
      bfr = __builtin_convertvector(bt, v16bf);
    }
#pragma unroll
    for (int mt = 0; mt < 4; ++mt) {
      v8bf lo = *(const v8bf*)&As[cur][mt * 16 + r][hi * 16];
      v8bf hp = *(const v8bf*)&As[cur][mt * 16 + r][hi * 16 + 8];
      v16bf a;
#pragma unroll
      for (int j = 0; j < 8; ++j) { a[j] = lo[j]; a[j + 8] = hp[j]; }
      acc[mt] = __builtin_amdgcn_wmma_f32_16x16x32_bf16(false, a, false, bfr,
                                                        (short)0, acc[mt], false, false);
    }
    __syncthreads();
  }

  const int n = nb + (wave << 4) + r;
  const float bv = bias[n];
#pragma unroll
  for (int mt = 0; mt < 4; ++mt) {
#pragma unroll
    for (int i = 0; i < 8; ++i) {
      const int m = m0 + mt * 16 + i + (hi << 3);
      out[(size_t)m * C_ + n] = acc[mt][i] + bv;
    }
  }
}

// ---------------------------------------------------------------------------
// Launch: ws layout = Q(16MB) | K(16MB) | V^T(16MB) | Y(16MB)  (64 MB total)
// ---------------------------------------------------------------------------
extern "C" void kernel_launch(void* const* d_in, const int* in_sizes, int n_in,
                              void* d_out, int out_size, void* d_ws, size_t ws_size,
                              hipStream_t stream) {
  const float* x      = (const float*)d_in[0];
  const float* w_attn = (const float*)d_in[1];
  const float* b_attn = (const float*)d_in[2];
  const float* w_proj = (const float*)d_in[3];
  const float* b_proj = (const float*)d_in[4];
  float* out = (float*)d_out;

  char* ws = (char*)d_ws;
  const size_t SEG = (size_t)B_ * H_ * T_ * D_ * sizeof(bf16s);  // 16 MB
  bf16s* qws  = (bf16s*)(ws + 0 * SEG);
  bf16s* kws  = (bf16s*)(ws + 1 * SEG);
  bf16s* vtws = (bf16s*)(ws + 2 * SEG);
  bf16s* yws  = (bf16s*)(ws + 3 * SEG);

  dim3 g1(N3_ / 128, M_ / 64);   // (24, 128)
  qkv_gemm_kernel<<<g1, 256, 0, stream>>>(x, w_attn, b_attn, qws, kws, vtws);

  const int nwaves = (B_ * H_ * (T_ / 16));   // 8192 waves, 8 per block
  attn_kernel<<<nwaves / 8, 256, 0, stream>>>(qws, kws, vtws, yws);

  dim3 g3(C_ / 128, M_ / 64);    // (8, 128)
  proj_gemm_kernel<<<g3, 256, 0, stream>>>(yws, w_proj, b_proj, out);
}
